// AttentionOp_34153579938058
// MI455X (gfx1250) — matrix operand order repaired and checked by reading, not compiled, not run on hardware
//
#include <hip/hip_runtime.h>

typedef __attribute__((ext_vector_type(16))) _Float16 v16h;
typedef __attribute__((ext_vector_type(8)))  float    v8f;

#define B_   2
#define T_   2048
#define NQ_  32
#define NKV_ 8
#define D_   128
#define BM   64
#define BN   64
#define NWAVE 4
#define LKP  (D_ + 8)   /* 136 halves per K row  */
#define LVP  (BN + 8)   /* 72 halves per Vt row  */
#define LPP  (BN + 8)   /* 72 halves per P row   */
#define MASKVAL (-0.7f * 3.4028234663852886e38f)

// pack two f32 -> packed f16 pair (single v_cvt_pk_rtz_f16_f32)
__device__ __forceinline__ unsigned pkh(float a, float b) {
  return __builtin_bit_cast(unsigned, __builtin_amdgcn_cvt_pkrtz(a, b));
}

__device__ __forceinline__ v8f wmma_f16(const unsigned a[8], const unsigned b[8], v8f c) {
  union { unsigned u[8]; v16h v; } ua, ub;
#pragma unroll
  for (int i = 0; i < 8; ++i) { ua.u[i] = a[i]; ub.u[i] = b[i]; }
  return __builtin_amdgcn_wmma_f32_16x16x32_f16(
      /*neg_a=*/false, ua.v, /*neg_b=*/false, ub.v,
      /*c_mod=*/(short)0, c, /*reuse_a=*/false, /*reuse_b=*/false);
}

__device__ __forceinline__ void ldfrag(unsigned dst[8], const unsigned short* p) {
  const unsigned* u = (const unsigned*)p;
#pragma unroll
  for (int i = 0; i < 8; ++i) dst[i] = u[i];
}

__global__ __launch_bounds__(128)
void fa_gqa_kernel(const float* __restrict__ Q, const float* __restrict__ K,
                   const float* __restrict__ V, const int* __restrict__ SEG,
                   float* __restrict__ O) {
  __shared__ __align__(16) unsigned short lK[BN][LKP];      // K tile, row-major f16
  __shared__ __align__(16) unsigned short lVt[D_][LVP];     // V tile, transposed f16
  __shared__ __align__(16) unsigned short lP[NWAVE][16][LPP];
  __shared__ int segQ[BM];
  __shared__ int segK[BN];

  const int tid  = threadIdx.x;
  const int wave = tid >> 5;
  const int lane = tid & 31;
  const int l16  = lane & 15;
  const int hi   = lane >> 4;

  const int q0  = blockIdx.x * BM;
  const int h   = blockIdx.y;
  const int b   = blockIdx.z;
  const int kvh = h >> 2;              // GQA: 4 q-heads per kv head
  const int qw  = q0 + wave * 16;

  if (tid < BM) segQ[tid] = SEG[b * T_ + q0 + tid];

  // ---- Q A-fragments (16x32 f16 per 32-wide D chunk), contiguous k = 16*hi + s order ----
  unsigned qf[4][8];
  {
    const float* qrow = Q + ((size_t)(b * T_ + qw + l16) * NQ_ + h) * D_;
#pragma unroll
    for (int c = 0; c < 4; ++c) {
      const float* p0 = qrow + 32 * c + 16 * hi;      // K = 16*hi + 0..7
      const float* p1 = qrow + 32 * c + 16 * hi + 8;  // K = 16*hi + 8..15
#pragma unroll
      for (int k = 0; k < 4; ++k) qf[c][k]     = pkh(p0[2*k], p0[2*k+1]);
#pragma unroll
      for (int k = 0; k < 4; ++k) qf[c][4 + k] = pkh(p1[2*k], p1[2*k+1]);
    }
  }

  v8f oacc[8];
#pragma unroll
  for (int dc = 0; dc < 8; ++dc)
#pragma unroll
    for (int r = 0; r < 8; ++r) oacc[dc][r] = 0.0f;

  float mrow[8], lrow[8];
#pragma unroll
  for (int r = 0; r < 8; ++r) { mrow[r] = MASKVAL; lrow[r] = 0.0f; }

  __syncthreads();                       // segQ visible
  const int qlo = segQ[wave * 16];       // sorted ids: endpoints bound the range
  const int qhi = segQ[wave * 16 + 15];

  const int ntiles = blockIdx.x + 1;     // causal: keys < q0+64, BN==BM
  for (int tile = 0; tile < ntiles; ++tile) {
    const int kv0 = tile * BN;

    __syncthreads();                     // previous tile's LDS fully consumed
    if (tid < BN) segK[tid] = SEG[b * T_ + kv0 + tid];
    // cooperative K/V tile load: 64 rows x 128 d, float4 per thread per iter
#pragma unroll
    for (int it = 0; it < 16; ++it) {
      int flat = it * 512 + tid * 4;
      int row = flat >> 7;
      int col = flat & 127;
      size_t gidx = ((size_t)(b * T_ + kv0 + row) * NKV_ + kvh) * D_ + col;
      float4 kq = *(const float4*)(K + gidx);
      float4 vq = *(const float4*)(V + gidx);
      unsigned* dst = (unsigned*)&lK[row][col];
      dst[0] = pkh(kq.x, kq.y);
      dst[1] = pkh(kq.z, kq.w);
      unsigned u0 = pkh(vq.x, vq.y);
      unsigned u1 = pkh(vq.z, vq.w);
      lVt[col + 0][row] = (unsigned short)u0;
      lVt[col + 1][row] = (unsigned short)(u0 >> 16);
      lVt[col + 2][row] = (unsigned short)u1;
      lVt[col + 3][row] = (unsigned short)(u1 >> 16);
      if (tile + 1 < ntiles) {           // prefetch next tile (global_prefetch_b8)
        __builtin_prefetch(K + gidx + (size_t)BN * NKV_ * D_, 0, 0);
        __builtin_prefetch(V + gidx + (size_t)BN * NKV_ * D_, 0, 0);
      }
    }
    __syncthreads();

    // ---- segment-overlap tile skip (sorted seg ids => endpoint tests) ----
    const int sk_lo = segK[0];
    const int sk_hi = segK[BN - 1];
    if (sk_lo > qhi || sk_hi < qlo) continue;   // tile contributes nothing

    // ---- S = Q x K^T : four 16x16 f32 tiles, pipelined B-fragment loads ----
    v8f S[4];
#pragma unroll
    for (int n = 0; n < 4; ++n)
#pragma unroll
      for (int r = 0; r < 8; ++r) S[n][r] = 0.0f;
    {
      unsigned kb[3][8];                 // rotating fragment buffers, 2 ahead
      ldfrag(kb[0], &lK[l16][16 * hi]);                 // s=0: (c=0,n=0)
      ldfrag(kb[1], &lK[16 + l16][16 * hi]);            // s=1: (c=0,n=1)
#pragma unroll
      for (int s = 0; s < 16; ++s) {
        const int c = s >> 2, n = s & 3;
        if (s + 2 < 16) {
          const int c2 = (s + 2) >> 2, n2 = (s + 2) & 3;
          ldfrag(kb[(s + 2) % 3], &lK[16 * n2 + l16][32 * c2 + 16 * hi]);
        }
        S[n] = wmma_f16(qf[c], kb[s % 3], S[n]);
      }
    }

    // ---- masking fixup only when needed (diagonal tile or seg boundary) ----
    const bool nomask = (tile != ntiles - 1) &&
                        (sk_lo == sk_hi) && (qlo == qhi) && (sk_lo == qlo);
    if (!nomask) {
      const int qs0 = segK[l16], qs1 = segK[16 + l16];
      const int qs2 = segK[32 + l16], qs3 = segK[48 + l16];
      const int c0 = kv0 + l16;
#pragma unroll
      for (int r = 0; r < 8; ++r) {
        const int m  = r + 8 * hi;
        const int qg = qw + m;
        const int qs = segQ[wave * 16 + m];
        if (!((c0      <= qg) && (qs0 == qs))) S[0][r] = MASKVAL;
        if (!((c0 + 16 <= qg) && (qs1 == qs))) S[1][r] = MASKVAL;
        if (!((c0 + 32 <= qg) && (qs2 == qs))) S[2][r] = MASKVAL;
        if (!((c0 + 48 <= qg) && (qs3 == qs))) S[3][r] = MASKVAL;
      }
    }

    // ---- online softmax (row r+8*hi lives in VGPR r, 16 lanes per row) ----
#pragma unroll
    for (int r = 0; r < 8; ++r) {
      const int m = r + 8 * hi;
      float s0 = S[0][r], s1 = S[1][r], s2 = S[2][r], s3 = S[3][r];
      float mx = fmaxf(fmaxf(s0, s1), fmaxf(s2, s3));
#pragma unroll
      for (int off = 8; off >= 1; off >>= 1)
        mx = fmaxf(mx, __shfl_xor(mx, off, 32));
      float mnew  = fmaxf(mrow[r], mx);
      float alpha = __expf(mrow[r] - mnew);
      float p0 = __expf(s0 - mnew);
      float p1 = __expf(s1 - mnew);
      float p2 = __expf(s2 - mnew);
      float p3 = __expf(s3 - mnew);
      float sum = (p0 + p1) + (p2 + p3);
#pragma unroll
      for (int off = 8; off >= 1; off >>= 1)
        sum += __shfl_xor(sum, off, 32);
      lrow[r] = lrow[r] * alpha + sum;
      mrow[r] = mnew;
#pragma unroll
      for (int dc = 0; dc < 8; ++dc) oacc[dc][r] *= alpha;
      unsigned u01 = pkh(p0, p1);
      unsigned u23 = pkh(p2, p3);
      lP[wave][m][l16]      = (unsigned short)u01;
      lP[wave][m][16 + l16] = (unsigned short)(u01 >> 16);
      lP[wave][m][32 + l16] = (unsigned short)u23;
      lP[wave][m][48 + l16] = (unsigned short)(u23 >> 16);
    }

    // ---- P A-fragments (two 16x32 f16: keys 0..31 and 32..63) ----
    unsigned pa0[8], pa1[8];
    {
      const unsigned* pu = (const unsigned*)&lP[wave][l16][0];
#pragma unroll
      for (int k = 0; k < 4; ++k) {
        pa0[k]     = pu[8 * hi + k];
        pa0[4 + k] = pu[8 * hi + 4 + k];
        pa1[k]     = pu[16 + 8 * hi + k];
        pa1[4 + k] = pu[20 + 8 * hi + k];
      }
    }
    // ---- O += P x V : sweep kg=0 over 8 D-chunks, then kg=1 (acc reuse 8 apart)
    {
      unsigned vb[3][8];                 // rotating fragment buffers, 2 ahead
      ldfrag(vb[0], &lVt[l16][16 * hi]);                // s=0: (kg=0,dc=0)
      ldfrag(vb[1], &lVt[16 + l16][16 * hi]);           // s=1: (kg=0,dc=1)
#pragma unroll
      for (int s = 0; s < 16; ++s) {
        const int kg = s >> 3, dc = s & 7;
        if (s + 2 < 16) {
          const int kg2 = (s + 2) >> 3, dc2 = (s + 2) & 7;
          ldfrag(vb[(s + 2) % 3], &lVt[dc2 * 16 + l16][32 * kg2 + 16 * hi]);
        }
        oacc[dc] = wmma_f16(kg ? pa1 : pa0, vb[s % 3], oacc[dc]);
      }
    }
  }

  // ---- normalize and store fp32 output ----
  float inv[8];
#pragma unroll
  for (int r = 0; r < 8; ++r) inv[r] = 1.0f / lrow[r];
#pragma unroll
  for (int dc = 0; dc < 8; ++dc) {
#pragma unroll
    for (int r = 0; r < 8; ++r) {
      int m = r + 8 * hi;
      int t = qw + m;
      int d = dc * 16 + l16;
      O[((size_t)(b * T_ + t) * NQ_ + h) * D_ + d] = oacc[dc][r] * inv[r];
    }
  }
}

extern "C" void kernel_launch(void* const* d_in, const int* in_sizes, int n_in,
                              void* d_out, int out_size, void* d_ws, size_t ws_size,
                              hipStream_t stream) {
  const float* Q   = (const float*)d_in[0];
  const float* K   = (const float*)d_in[1];
  const float* V   = (const float*)d_in[2];
  const int*   SEG = (const int*)d_in[3];
  float* O = (float*)d_out;
  dim3 grid(T_ / BM, NQ_, B_);   // (32 q-tiles, 32 heads, 2 batches)
  fa_gqa_kernel<<<grid, 128, 0, stream>>>(Q, K, V, SEG, O);
}
